// VectorQuantizer_13297218748621
// MI455X (gfx1250) — matrix operand order repaired and checked by reading, not compiled, not run on hardware
//
#include <hip/hip_runtime.h>
#include <hip/hip_bf16.h>

// ---------------------------------------------------------------------------
// VQ-VAE quantizer for MI455X (gfx1250, wave32, WMMA)
//   z: [16,256,64,64] f32, codebook: [1024,256] f32
//   Distance GEMM (65536x1024x256) via compensated-bf16 WMMA:
//     x.e ~= xh.eh + xh.el + xl.eh   (3x v_wmma_f32_16x16x32_bf16 per k-step)
//   argmin uses  cn[e] - 2*x.e  (||x||^2 constant per pixel).
// Outputs (concatenated in d_out, all f32):
//   [0 .. 16777215]  z_q_out  [16,256,64,64]
//   [16777216]       codebook_loss = 1.25 * SSE / (N*C)
//   [16777217 .. ]   indices  (65536, stored as float)
// ---------------------------------------------------------------------------

typedef __attribute__((ext_vector_type(16))) __bf16 v16bf;
typedef __attribute__((ext_vector_type(8)))  float  v8f;

union FragAB { v16bf v; uint4 q[2]; };

__device__ __forceinline__ unsigned short bf16_rne(float x) {
    unsigned u = __float_as_uint(x);
    unsigned r = u + 0x7FFFu + ((u >> 16) & 1u);
    return (unsigned short)(r >> 16);
}
__device__ __forceinline__ float bf16f(unsigned short h) {
    return __uint_as_float(((unsigned)h) << 16);
}

#define NCODE   1024
#define CDIM    256
#define HW      4096      // 64*64
#define PTILE   32        // pixels per workgroup (one half-row of W)
#define ROWB    528       // LDS row bytes: 256 bf16 (512B) + 16B pad (bank spread)
#define NWG     2048      // 65536 / PTILE

// --- Prep: split codebook into hi/lo bf16 (row major) + code norms ----------
__global__ __launch_bounds__(256) void vq_prep(const float* __restrict__ cb,
                                               unsigned short* __restrict__ cb_hi,
                                               unsigned short* __restrict__ cb_lo,
                                               float* __restrict__ cn) {
    __shared__ float red[256];
    unsigned code = blockIdx.x;           // 1024 blocks
    unsigned t = threadIdx.x;             // 256 threads = one k each
    float v = cb[(size_t)code * CDIM + t];
    unsigned short hh = bf16_rne(v);
    unsigned short ll = bf16_rne(v - bf16f(hh));
    cb_hi[(size_t)code * CDIM + t] = hh;
    cb_lo[(size_t)code * CDIM + t] = ll;
    red[t] = v * v;
    __syncthreads();
    for (int s = 128; s > 0; s >>= 1) {
        if (t < (unsigned)s) red[t] += red[t + s];
        __syncthreads();
    }
    if (t == 0) cn[code] = red[0];
}

// --- Main: distances + argmin + gather + partial loss -----------------------
__global__ __launch_bounds__(64) void vq_main(const float* __restrict__ z,
                                              const float* __restrict__ cb,
                                              const unsigned short* __restrict__ cb_hi,
                                              const unsigned short* __restrict__ cb_lo,
                                              const float* __restrict__ cn,
                                              float* __restrict__ out_q,
                                              float* __restrict__ out_idx,
                                              float* __restrict__ partials) {
    __shared__ unsigned char lds[2 * PTILE * ROWB];   // [hi rows][lo rows]
    __shared__ float red[64];

    unsigned g = blockIdx.x;
    unsigned base_n = g * PTILE;
    unsigned b   = base_n >> 12;          // /4096
    unsigned rem = base_n & 4095u;
    unsigned h   = rem >> 6;
    unsigned w0  = rem & 63u;             // 0 or 32
    size_t zbase = (size_t)b * CDIM * HW + (size_t)h * 64 + w0;  // + c*4096 + w'

    unsigned t    = threadIdx.x;
    unsigned lane = t & 31u;
    unsigned wave = t >> 5;               // 0..1

    // ---- Stage X tile to LDS as hi/lo bf16, [pixel][k] with padded rows ----
    {
        unsigned wpix = t & 31u;          // pixel within tile
        unsigned half = t >> 5;           // channel-pair selector
        #pragma unroll 4
        for (int i = 0; i < 64; ++i) {
            unsigned c = (unsigned)i * 4u + half * 2u;
            float v0 = z[zbase + (size_t)c * HW + wpix];
            float v1 = z[zbase + (size_t)(c + 1) * HW + wpix];
            unsigned short h0 = bf16_rne(v0), h1 = bf16_rne(v1);
            unsigned short l0 = bf16_rne(v0 - bf16f(h0));
            unsigned short l1 = bf16_rne(v1 - bf16f(h1));
            *(unsigned*)(lds + wpix * ROWB + c * 2) =
                (unsigned)h0 | ((unsigned)h1 << 16);
            *(unsigned*)(lds + PTILE * ROWB + wpix * ROWB + c * 2) =
                (unsigned)l0 | ((unsigned)l1 << 16);
        }
    }
    __syncthreads();

    // ---- Per-wave: 16 pixels x 1024 codes ----------------------------------
    unsigned laneHi = lane >> 4;                        // 0 | 1
    unsigned pix    = (wave << 4) | (lane & 15u);       // local pixel 0..31
    unsigned koffA  = laneHi * 8;                       // A-fragment K sub-offset

    float    minv = 3.4e38f;
    unsigned mini = 0u;

    for (int tile = 0; tile < 64; ++tile) {
        const unsigned short* arow_h = cb_hi + (size_t)(tile * 16 + (lane & 15u)) * CDIM;
        const unsigned short* arow_l = cb_lo + (size_t)(tile * 16 + (lane & 15u)) * CDIM;
        v8f acc = {};
        #pragma unroll
        for (int ks = 0; ks < 8; ++ks) {
            int k0 = ks * 32;
            FragAB ah, al, bh, bl;
            // A (codes, 16x32): lanes<16 K{0-7,16-23}, lanes>=16 K{8-15,24-31}
            ah.q[0] = *(const uint4*)(arow_h + k0 + koffA);
            ah.q[1] = *(const uint4*)(arow_h + k0 + 16 + koffA);
            al.q[0] = *(const uint4*)(arow_l + k0 + koffA);
            al.q[1] = *(const uint4*)(arow_l + k0 + 16 + koffA);
            // B (pixels, 32x16): lanes<16 K{0-7,16-23}, lanes>=16 K{8-15,24-31}
            const unsigned char* brh = lds + pix * ROWB + (k0 + laneHi * 8) * 2;
            const unsigned char* brl = brh + PTILE * ROWB;
            bh.q[0] = *(const uint4*)(brh);
            bh.q[1] = *(const uint4*)(brh + 32);
            bl.q[0] = *(const uint4*)(brl);
            bl.q[1] = *(const uint4*)(brl + 32);
            acc = __builtin_amdgcn_wmma_f32_16x16x32_bf16(false, ah.v, false, bh.v,
                                                          (short)0, acc, false, false);
            acc = __builtin_amdgcn_wmma_f32_16x16x32_bf16(false, ah.v, false, bl.v,
                                                          (short)0, acc, false, false);
            acc = __builtin_amdgcn_wmma_f32_16x16x32_bf16(false, al.v, false, bh.v,
                                                          (short)0, acc, false, false);
        }
        // D layout: VGPR r holds code M = r + 8*laneHi for this lane's pixel.
        unsigned cbase = (unsigned)tile * 16u + laneHi * 8u;
        float4 c0 = *(const float4*)(cn + cbase);
        float4 c1 = *(const float4*)(cn + cbase + 4);
        float cn8[8] = {c0.x, c0.y, c0.z, c0.w, c1.x, c1.y, c1.z, c1.w};
        #pragma unroll
        for (int r = 0; r < 8; ++r) {
            float d = cn8[r] - 2.0f * acc[r];
            if (d < minv) { minv = d; mini = cbase + (unsigned)r; }
        }
    }

    // Combine lane halves (same pixel lives in lane L and L^16); keep lowest idx on tie.
    float    ov = __shfl_xor(minv, 16, 32);
    unsigned oi = (unsigned)__shfl_xor((int)mini, 16, 32);
    if (ov < minv || (ov == minv && oi < mini)) { minv = ov; mini = oi; }

    // ---- Outputs: indices, gathered z_q, partial SSE -----------------------
    if (lane < 16u) out_idx[base_n + pix] = (float)mini;

    size_t obase = (size_t)b * CDIM * HW + (size_t)h * 64 + (w0 + pix); // + c*4096
    const float* crow = cb + (size_t)mini * CDIM;
    float sse = 0.0f;
    #pragma unroll 4
    for (int c0 = 0; c0 < CDIM; c0 += 2) {
        unsigned c = (unsigned)c0 + laneHi;
        float  qv   = crow[c];
        size_t addr = obase + (size_t)c * HW;
        float  zv   = z[addr];        // L2-hot (all of z fits in 192MB L2)
        out_q[addr] = qv;
        float e = qv - zv;
        sse += e * e;
    }

    red[t] = sse;
    __syncthreads();
    if (t == 0) {
        float s = 0.0f;
        for (int i = 0; i < 64; ++i) s += red[i];   // fixed order: deterministic
        partials[g] = s;
    }
}

// --- Final deterministic loss reduction ------------------------------------
__global__ __launch_bounds__(256) void vq_finish(const float* __restrict__ partials,
                                                 float* __restrict__ out_loss) {
    __shared__ float red[256];
    unsigned t = threadIdx.x;
    float s = 0.0f;
    for (int i = 0; i < NWG / 256; ++i) s += partials[t + (unsigned)i * 256u];
    red[t] = s;
    __syncthreads();
    if (t == 0) {
        float tot = 0.0f;
        for (int i = 0; i < 256; ++i) tot += red[i];
        // codebook_loss = (1 + BETA) * SSE / (N*C), BETA = 0.25
        *out_loss = tot * (1.25f / 16777216.0f);
    }
}

extern "C" void kernel_launch(void* const* d_in, const int* in_sizes, int n_in,
                              void* d_out, int out_size, void* d_ws, size_t ws_size,
                              hipStream_t stream) {
    const float* z  = (const float*)d_in[0];   // [16,256,64,64]
    const float* cb = (const float*)d_in[1];   // [1024,256]

    float* out_q    = (float*)d_out;           // 16,777,216
    float* out_loss = out_q + 16777216;        // 1
    float* out_idx  = out_loss + 1;            // 65,536

    unsigned char* ws = (unsigned char*)d_ws;
    unsigned short* cb_hi = (unsigned short*)(ws);                 // 512 KB
    unsigned short* cb_lo = (unsigned short*)(ws + 524288);        // 512 KB
    float*          cn    = (float*)(ws + 1048576);                // 4 KB
    float*          parts = (float*)(ws + 1048576 + 4096);         // 8 KB

    vq_prep  <<<NCODE, 256, 0, stream>>>(cb, cb_hi, cb_lo, cn);
    vq_main  <<<NWG,    64, 0, stream>>>(z, cb, cb_hi, cb_lo, cn,
                                         out_q, out_idx, parts);
    vq_finish<<<1,     256, 0, stream>>>(parts, out_loss);
}
